// ModulatorAttention_22978075034424
// MI455X (gfx1250) — matrix operand order repaired and checked by reading, not compiled, not run on hardware
//
#include <hip/hip_runtime.h>
#include <cstdint>

// ---------------------------------------------------------------------------
// ModulatorAttention for MI455X (gfx1250, wave32, WMMA).
// B=2, N=2048, C=1024, H=16, D=64. All matmuls via v_wmma_f32_16x16x32_bf16.
// GEMM B-tiles staged through LDS with GLOBAL_LOAD_ASYNC_TO_LDS_B128
// (double-buffered, ASYNCcnt-tracked); working set is L2-resident (192MB).
// ---------------------------------------------------------------------------

typedef __bf16 bf16;
typedef __attribute__((ext_vector_type(8)))  __bf16 bf16x8;   // 16 bytes
typedef __attribute__((ext_vector_type(16))) __bf16 bf16x16;  // 32 bytes
typedef __attribute__((ext_vector_type(8)))  float  f32x8;

#define WMMA_BF16(a, b, c) \
  __builtin_amdgcn_wmma_f32_16x16x32_bf16(false, (a), false, (b), (short)0, (c), false, false)

static constexpr int BATCH = 2;
static constexpr int SEQ   = 2048;
static constexpr int CH    = 1024;
static constexpr int HEADS = 16;
static constexpr int HDIM  = 64;
static constexpr int TOK   = BATCH * SEQ;   // 4096
static constexpr int BPAD  = 40;            // padded LDS row (80B): conflict-free 16B reads

__device__ inline bf16x16 pack16(bf16x8 lo, bf16x8 hi) {
  bf16x16 r;
#pragma unroll
  for (int i = 0; i < 8; ++i) { r[i] = lo[i]; r[8 + i] = hi[i]; }
  return r;
}

// CDNA5 async copy: global -> LDS, 16B per lane, tracked by ASYNCcnt.
__device__ inline void async_load_b128(uint32_t lds_off, const void* gaddr) {
  asm volatile("global_load_async_to_lds_b128 %0, %1, off"
               :: "v"(lds_off), "v"((uint64_t)(uintptr_t)gaddr)
               : "memory");
}
__device__ inline void wait_async0() {
  asm volatile("s_wait_asynccnt 0x0" ::: "memory");
}

// ---------------------------------------------------------------------------
// fp32 -> bf16 elementwise convert
// ---------------------------------------------------------------------------
__global__ __launch_bounds__(256)
void cvt_f32_bf16_kernel(const float* __restrict__ in, bf16* __restrict__ out, int n) {
  int i = blockIdx.x * 256 + threadIdx.x;
  if (i < n) out[i] = (bf16)in[i];
}

// ---------------------------------------------------------------------------
// C[M,N] = A[M,K] * Bt[N,K]^T + bias[N]   (bf16 in, f32 accumulate/out)
// Block = 8 waves (32x8): block tile 256(M) x 64(N); wave tile 32x64.
// B tile (64x32 bf16, shared by all waves) is staged via async-to-LDS,
// double-buffered; waves read B fragments with ds_load_b128.
// A fragments come straight from global (per-wave-unique rows, L2-resident).
// ---------------------------------------------------------------------------
__global__ __launch_bounds__(256)
void gemm_bf16_kernel(const bf16* __restrict__ A, const bf16* __restrict__ Bt,
                      const float* __restrict__ bias, float* __restrict__ C,
                      int M, int N, int K) {
  __shared__ bf16 ldsB[2][64][BPAD];

  const int lane  = threadIdx.x;        // 0..31
  const int wid   = threadIdx.y;        // 0..7
  const int tid   = wid * 32 + lane;    // 0..255
  const int col   = lane & 15;
  const int half  = lane >> 4;
  const int mBase = blockIdx.y * 256 + wid * 32;
  const int nBase = blockIdx.x * 64;

  // staging map: thread -> (row sn, 8-elem chunk sk) of the 64x32 B tile
  const int sn = tid >> 2;              // 0..63
  const int sk = (tid & 3) * 8;         // 0,8,16,24
  const bf16* gB = Bt + (size_t)(nBase + sn) * K + sk;

  f32x8 acc[8] = {};

  const bf16* aRow0 = A + (size_t)(mBase + col) * K + half * 8;
  const bf16* aRow1 = aRow0 + (size_t)16 * K;

  // prefetch first B tile into buffer 0
  async_load_b128((uint32_t)(uintptr_t)&ldsB[0][sn][sk], gB);
  wait_async0();
  __syncthreads();

  for (int k = 0; k < K; k += 32) {
    const int cur = (k >> 5) & 1;
    if (k + 32 < K)
      async_load_b128((uint32_t)(uintptr_t)&ldsB[cur ^ 1][sn][sk], gB + k + 32);

    // A fragments (16x32 each) for both 16-row halves of the wave tile
    bf16x16 a0 = pack16(*(const bf16x8*)(aRow0 + k), *(const bf16x8*)(aRow0 + k + 16));
    bf16x16 a1 = pack16(*(const bf16x8*)(aRow1 + k), *(const bf16x8*)(aRow1 + k + 16));

#pragma unroll
    for (int t = 0; t < 4; ++t) {
      const bf16* bp = &ldsB[cur][16 * t + col][half * 8];
      bf16x16 bfrag = pack16(*(const bf16x8*)bp, *(const bf16x8*)(bp + 16));
      acc[t]     = WMMA_BF16(a0, bfrag, acc[t]);
      acc[4 + t] = WMMA_BF16(a1, bfrag, acc[4 + t]);
    }

    wait_async0();     // next tile's DMA (this wave's share) complete
    __syncthreads();   // publish across waves; orders this iter's ds reads too
  }

#pragma unroll
  for (int t = 0; t < 4; ++t) {
    const int n = nBase + 16 * t + col;
    const float bv = bias[n];
#pragma unroll
    for (int mh = 0; mh < 2; ++mh) {
#pragma unroll
      for (int j = 0; j < 8; ++j) {
        const int m = mBase + mh * 16 + half * 8 + j;
        C[(size_t)m * N + n] = acc[mh * 4 + t][j] + bv;
      }
    }
  }
}

// ---------------------------------------------------------------------------
// Split QKV [TOK, 3C] f32 -> Q,K [B,H,N,D] bf16 (RMS-normed), V^T [B,H,D,N] bf16.
// One wave per (b,n,h); lane owns d=lane and d=lane+32.
// ---------------------------------------------------------------------------
__global__ __launch_bounds__(256)
void qkv_norm_split_kernel(const float* __restrict__ qkv,
                           const float* __restrict__ qw, const float* __restrict__ kw,
                           bf16* __restrict__ qb, bf16* __restrict__ kb,
                           bf16* __restrict__ vt) {
  const int lane = threadIdx.x & 31;
  const int wid  = threadIdx.x >> 5;
  const int wg   = blockIdx.x * 8 + wid;          // 0 .. B*N*H-1
  const int h    = wg & 15;
  const int n    = (wg >> 4) & (SEQ - 1);
  const int b    = wg >> 15;

  const float* base = qkv + (size_t)(b * SEQ + n) * (3 * CH) + h * HDIM;
  const size_t rowOff = ((size_t)(b * HEADS + h) * SEQ + n) * HDIM;

  // ---- Q ----
  {
    float a = base[lane], c = base[lane + 32];
    float ss = a * a + c * c;
#pragma unroll
    for (int off = 1; off < 32; off <<= 1) ss += __shfl_xor(ss, off, 32);
    float r = __frsqrt_rn(ss * (1.0f / HDIM) + 1e-6f);
    qb[rowOff + lane]      = (bf16)(a * r * qw[lane]);
    qb[rowOff + lane + 32] = (bf16)(c * r * qw[lane + 32]);
  }
  // ---- K ----
  {
    float a = base[CH + lane], c = base[CH + lane + 32];
    float ss = a * a + c * c;
#pragma unroll
    for (int off = 1; off < 32; off <<= 1) ss += __shfl_xor(ss, off, 32);
    float r = __frsqrt_rn(ss * (1.0f / HDIM) + 1e-6f);
    kb[rowOff + lane]      = (bf16)(a * r * kw[lane]);
    kb[rowOff + lane + 32] = (bf16)(c * r * kw[lane + 32]);
  }
  // ---- V (transposed: [b,h][d][n]) ----
  {
    float a = base[2 * CH + lane], c = base[2 * CH + lane + 32];
    const size_t vBase = ((size_t)(b * HEADS + h) * HDIM) * SEQ + n;
    vt[vBase + (size_t)lane * SEQ]        = (bf16)a;
    vt[vBase + (size_t)(lane + 32) * SEQ] = (bf16)c;
  }
}

// ---------------------------------------------------------------------------
// Flash attention: per (b,h), each wave owns 16 query rows, online softmax
// over key chunks of 32. P transposed to A-frag layout through padded LDS.
// Output written as bf16 straight into [B,N,C].
// ---------------------------------------------------------------------------
__global__ __launch_bounds__(128)
void attn_kernel(const bf16* __restrict__ Q, const bf16* __restrict__ Kmat,
                 const bf16* __restrict__ Vt, bf16* __restrict__ Out) {
  __shared__ __bf16 plds[4][16][BPAD];   // per-wave P staging (padded rows)

  const int lane = threadIdx.x & 31;
  const int wid  = threadIdx.x >> 5;
  const int col  = lane & 15;
  const int half = lane >> 4;

  const int bh = blockIdx.y;           // 0..31
  const int b  = bh >> 4, h = bh & 15;
  const int qBase = blockIdx.x * 64 + wid * 16;

  const bf16* Qh = Q    + (size_t)bh * SEQ * HDIM;
  const bf16* Kh = Kmat + (size_t)bh * SEQ * HDIM;
  const bf16* Vh = Vt   + (size_t)bh * HDIM * SEQ;

  // Q A-frags for d=[0,32) and d=[32,64), loaded once
  bf16x16 aq[2];
  {
    const bf16* qRow = Qh + (size_t)(qBase + col) * HDIM + half * 16;
#pragma unroll
    for (int s = 0; s < 2; ++s)
      aq[s] = pack16(*(const bf16x8*)(qRow + 32 * s),
                     *(const bf16x8*)(qRow + 32 * s + 8));
  }

  f32x8 o[4] = {};
  float mrow[8], lrow[8];
#pragma unroll
  for (int j = 0; j < 8; ++j) { mrow[j] = -INFINITY; lrow[j] = 0.0f; }

  const float sscale = 0.125f * 1.44269504088896340736f;  // (1/sqrt(D)) * log2(e)

  for (int k0 = 0; k0 < SEQ; k0 += 32) {
    // ---- S = Q * K^T (two 16x16 tiles over 32 keys) ----
    f32x8 s[2] = {};
#pragma unroll
    for (int t = 0; t < 2; ++t) {
      const bf16* kPtr = Kh + (size_t)(k0 + 16 * t + col) * HDIM + half * 16;
      bf16x16 b0 = *(const bf16x16*)kPtr;          // d in [0,32)
      bf16x16 b1 = *(const bf16x16*)(kPtr + 32);   // d in [32,64)
      s[t] = WMMA_BF16(aq[0], b0, s[t]);
      s[t] = WMMA_BF16(aq[1], b1, s[t]);
    }

    // ---- online softmax (log2 domain), P -> LDS as bf16 ----
#pragma unroll
    for (int j = 0; j < 8; ++j) {
      float v0 = s[0][j] * sscale;
      float v1 = s[1][j] * sscale;
      float mx = fmaxf(v0, v1);
#pragma unroll
      for (int off = 1; off < 16; off <<= 1) mx = fmaxf(mx, __shfl_xor(mx, off, 32));
      float mnew = fmaxf(mrow[j], mx);
      float corr = __builtin_exp2f(mrow[j] - mnew);
      mrow[j] = mnew;
      float p0 = __builtin_exp2f(v0 - mnew);
      float p1 = __builtin_exp2f(v1 - mnew);
      float ps = p0 + p1;
#pragma unroll
      for (int off = 1; off < 16; off <<= 1) ps += __shfl_xor(ps, off, 32);
      lrow[j] = lrow[j] * corr + ps;
#pragma unroll
      for (int t = 0; t < 4; ++t) o[t][j] *= corr;
      const int row = half * 8 + j;
      plds[wid][row][col]      = (bf16)p0;
      plds[wid][row][16 + col] = (bf16)p1;
    }
    __syncthreads();

    // ---- P as A-frag (16 queries x 32 keys) ----
    bf16x16 ap;
    {
      const __bf16* pRow = &plds[wid][col][0];
      ap = pack16(*(const bf16x8*)(pRow + half * 16),
                  *(const bf16x8*)(pRow + half * 16 + 8));
    }
    __syncthreads();

    // ---- O += P * V  (B-frags from V^T: contiguous keys per d column) ----
#pragma unroll
    for (int t = 0; t < 4; ++t) {
      const bf16* vPtr = Vh + (size_t)(16 * t + col) * SEQ + k0 + half * 16;
      bf16x16 bv = *(const bf16x16*)vPtr;
      o[t] = WMMA_BF16(ap, bv, o[t]);
    }
  }

  // ---- normalize and write [B,N,C] bf16 ----
#pragma unroll
  for (int t = 0; t < 4; ++t) {
#pragma unroll
    for (int j = 0; j < 8; ++j) {
      const int m = qBase + half * 8 + j;
      const int d = 16 * t + col;
      Out[(size_t)(b * SEQ + m) * CH + h * HDIM + d] = (bf16)(o[t][j] / lrow[j]);
    }
  }
}

// ---------------------------------------------------------------------------
// launch
// ---------------------------------------------------------------------------
extern "C" void kernel_launch(void* const* d_in, const int* in_sizes, int n_in,
                              void* d_out, int out_size, void* d_ws, size_t ws_size,
                              hipStream_t stream) {
  const float* x      = (const float*)d_in[0];  // [B,N,C]
  const float* w_qkv  = (const float*)d_in[1];  // [3C,C]
  const float* b_qkv  = (const float*)d_in[2];  // [3C]
  const float* qnw    = (const float*)d_in[3];  // [D]
  const float* knw    = (const float*)d_in[4];  // [D]
  const float* w_proj = (const float*)d_in[5];  // [C,C]
  const float* b_proj = (const float*)d_in[6];  // [C]
  float* out = (float*)d_out;

  char* ws = (char*)d_ws;
  size_t off = 0;
  bf16*  xb    = (bf16*)(ws + off); off += (size_t)TOK * CH * 2;            // 8 MB
  bf16*  wqkvb = (bf16*)(ws + off); off += (size_t)3 * CH * CH * 2;         // 6 MB
  bf16*  wprjb = (bf16*)(ws + off); off += (size_t)CH * CH * 2;             // 2 MB
  float* qkvf  = (float*)(ws + off); off += (size_t)TOK * 3 * CH * 4;       // 48 MB
  bf16*  qb    = (bf16*)(ws + off); off += (size_t)TOK * CH * 2;            // 8 MB
  bf16*  kb    = (bf16*)(ws + off); off += (size_t)TOK * CH * 2;            // 8 MB
  bf16*  vtb   = (bf16*)(ws + off); off += (size_t)TOK * CH * 2;            // 8 MB
  bf16*  attnb = (bf16*)(ws + off); off += (size_t)TOK * CH * 2;            // 8 MB

  // 1) converts
  {
    int n0 = TOK * CH;          // x
    int n1 = 3 * CH * CH;       // w_qkv
    int n2 = CH * CH;           // w_proj
    cvt_f32_bf16_kernel<<<(n0 + 255) / 256, 256, 0, stream>>>(x, xb, n0);
    cvt_f32_bf16_kernel<<<(n1 + 255) / 256, 256, 0, stream>>>(w_qkv, wqkvb, n1);
    cvt_f32_bf16_kernel<<<(n2 + 255) / 256, 256, 0, stream>>>(w_proj, wprjb, n2);
  }

  // 2) QKV GEMM: [4096,1024] x [3072,1024]^T -> [4096,3072] f32 (+bias)
  gemm_bf16_kernel<<<dim3(3 * CH / 64, TOK / 256), dim3(32, 8), 0, stream>>>(
      xb, wqkvb, b_qkv, qkvf, TOK, 3 * CH, CH);

  // 3) split + RMSNorm -> Q,K [B,H,N,D] bf16 ; V^T [B,H,D,N] bf16
  qkv_norm_split_kernel<<<(BATCH * SEQ * HEADS) / 8, 256, 0, stream>>>(
      qkvf, qnw, knw, qb, kb, vtb);

  // 4) flash attention -> [B,N,C] bf16
  attn_kernel<<<dim3(SEQ / 64, BATCH * HEADS), 128, 0, stream>>>(qb, kb, vtb, attnb);

  // 5) projection GEMM: [4096,1024] x [1024,1024]^T -> d_out f32 (+bias)
  gemm_bf16_kernel<<<dim3(CH / 64, TOK / 256), dim3(32, 8), 0, stream>>>(
      attnb, wprjb, b_proj, out, TOK, CH, CH);

  (void)in_sizes; (void)n_in; (void)out_size; (void)ws_size;
}
